// Point_MAE_Mamba_56951266345007
// MI455X (gfx1250) — compile-verified, matrix operand order checked by reading, not run on hardware
//
#include <hip/hip_runtime.h>
#include <cstdint>
#include <cstddef>

// ---------------- model dims ----------------
#define Bb    16
#define Np    2048
#define Gg    128
#define GSs   32
#define DM    384
#define DI    768
#define DSt   16
#define DTR   24
#define ENCD  12
#define DECD  4
#define NMASK 76
#define NVIS  52

typedef __attribute__((ext_vector_type(16))) _Float16 v16h;
typedef __attribute__((ext_vector_type(8)))  _Float16 v8h;
typedef __attribute__((ext_vector_type(8)))  float    v8f;

#define CEILDIV(a,b) (((a)+(b)-1)/(b))

__device__ __forceinline__ float sigf(float x){ return 1.f/(1.f+expf(-x)); }

__device__ __forceinline__ float activate(float x, int act){
  if (act == 1) return x > 0.f ? x : 0.f;                       // relu
  if (act == 2) return 0.5f*x*(1.f+erff(x*0.70710678118f));     // gelu (exact)
  if (act == 3) return x > 20.f ? x : log1pf(expf(x));          // softplus
  return x;
}

// ---------------------------------------------------------------------------
// Fast-path WMMA GEMM (requires K % 32 == 0):
//   C[M,N] = act( A[M,K](lda) @ W[N,K]^T + bias[N] )
// Block tile 64x128, 8 waves, each wave owns a 32x32 C tile -> 4 WMMA per
// 32-wide K step.  Staging: float4 global loads (row-guard only, no
// per-element divergence), f32->f16 in registers, 128-bit LDS stores.
// ---------------------------------------------------------------------------
__global__ __launch_bounds__(256) void gemm_wmma_aligned_kernel(
    const float* __restrict__ Ag, const float* __restrict__ Wg,
    const float* __restrict__ bias, float* __restrict__ Cg,
    int M, int Nn, int K, int lda, int act)
{
  __shared__ __align__(16) _Float16 sA[64*32];
  __shared__ __align__(16) _Float16 sB[128*32];
  const int tid  = threadIdx.x;
  const int wave = tid >> 5;
  const int lane = tid & 31;
  const int bm = blockIdx.x * 64;
  const int bn = blockIdx.y * 128;
  const int wm = (wave & 1) * 32;   // wave row offset in tile
  const int wn = (wave >> 1) * 32;  // wave col offset in tile

  v8f acc00 = {0.f,0.f,0.f,0.f,0.f,0.f,0.f,0.f};
  v8f acc01 = acc00, acc10 = acc00, acc11 = acc00;

  // staging assignments
  const int ar = tid >> 2, ac = (tid & 3) * 8;     // A: 64 rows x 32 cols, 8 elems/thread
  const int br = tid >> 1, bc = (tid & 1) * 16;    // B: 128 rows x 32 cols, 16 elems/thread
  const int garow = bm + ar;
  const int gbrow = bn + br;
  const float4 z4 = {0.f,0.f,0.f,0.f};

  for (int k0 = 0; k0 < K; k0 += 32) {
    float4 a0 = z4, a1 = z4;
    if (garow < M) {
      const float* p = Ag + (size_t)garow*lda + k0 + ac;
      a0 = *(const float4*)p;
      a1 = *(const float4*)(p + 4);
    }
    float4 b0 = z4, b1 = z4, b2 = z4, b3 = z4;
    if (gbrow < Nn) {
      const float* p = Wg + (size_t)gbrow*K + k0 + bc;
      b0 = *(const float4*)p;       b1 = *(const float4*)(p + 4);
      b2 = *(const float4*)(p + 8); b3 = *(const float4*)(p + 12);
    }
    v8h ha;
    ha[0]=(_Float16)a0.x; ha[1]=(_Float16)a0.y; ha[2]=(_Float16)a0.z; ha[3]=(_Float16)a0.w;
    ha[4]=(_Float16)a1.x; ha[5]=(_Float16)a1.y; ha[6]=(_Float16)a1.z; ha[7]=(_Float16)a1.w;
    *(v8h*)&sA[ar*32 + ac] = ha;
    v8h hb0, hb1;
    hb0[0]=(_Float16)b0.x; hb0[1]=(_Float16)b0.y; hb0[2]=(_Float16)b0.z; hb0[3]=(_Float16)b0.w;
    hb0[4]=(_Float16)b1.x; hb0[5]=(_Float16)b1.y; hb0[6]=(_Float16)b1.z; hb0[7]=(_Float16)b1.w;
    hb1[0]=(_Float16)b2.x; hb1[1]=(_Float16)b2.y; hb1[2]=(_Float16)b2.z; hb1[3]=(_Float16)b2.w;
    hb1[4]=(_Float16)b3.x; hb1[5]=(_Float16)b3.y; hb1[6]=(_Float16)b3.z; hb1[7]=(_Float16)b3.w;
    *(v8h*)&sB[br*32 + bc]     = hb0;
    *(v8h*)&sB[br*32 + bc + 8] = hb1;
    __syncthreads();

    // Fragments per ISA 7.12.2 16-bit layout:
    // lanes 0-15 hold K = 0..7,16..23 ; lanes 16-31 hold K = 8..15,24..31
    v16h af0, af1, bf0, bf1;
    const int half8 = (lane >> 4) << 3;
    const int l16   = lane & 15;
    #pragma unroll
    for (int j = 0; j < 16; ++j) {
      int kk = half8 + (j & 7) + ((j >> 3) << 4);
      af0[j] = sA[(wm      + l16)*32 + kk];
      af1[j] = sA[(wm + 16 + l16)*32 + kk];
      bf0[j] = sB[(wn      + l16)*32 + kk];
      bf1[j] = sB[(wn + 16 + l16)*32 + kk];
    }
    acc00 = __builtin_amdgcn_wmma_f32_16x16x32_f16(false, af0, false, bf0, (short)0, acc00, false, false);
    acc01 = __builtin_amdgcn_wmma_f32_16x16x32_f16(false, af0, false, bf1, (short)0, acc01, false, false);
    acc10 = __builtin_amdgcn_wmma_f32_16x16x32_f16(false, af1, false, bf0, (short)0, acc10, false, false);
    acc11 = __builtin_amdgcn_wmma_f32_16x16x32_f16(false, af1, false, bf1, (short)0, acc11, false, false);
    __syncthreads();
  }
  // C/D layout: VGPR r -> row (r + 8*(lane>=16)), col = lane&15
  const int rb0  = bm + wm + ((lane >> 4) << 3);
  const int rb1  = rb0 + 16;
  const int col0 = bn + wn + (lane & 15);
  const int col1 = col0 + 16;
  float bv0 = 0.f, bv1 = 0.f;
  if (bias) { if (col0 < Nn) bv0 = bias[col0]; if (col1 < Nn) bv1 = bias[col1]; }
  #pragma unroll
  for (int r = 0; r < 8; ++r) {
    int row = rb0 + r;
    if (row < M) {
      if (col0 < Nn) Cg[(size_t)row*Nn + col0] = activate(acc00[r] + bv0, act);
      if (col1 < Nn) Cg[(size_t)row*Nn + col1] = activate(acc01[r] + bv1, act);
    }
    int row2 = rb1 + r;
    if (row2 < M) {
      if (col0 < Nn) Cg[(size_t)row2*Nn + col0] = activate(acc10[r] + bv0, act);
      if (col1 < Nn) Cg[(size_t)row2*Nn + col1] = activate(acc11[r] + bv1, act);
    }
  }
}

// ---------------------------------------------------------------------------
// Generic WMMA GEMM fallback (ragged K: 3, 24).  Tiny share of total FLOPs.
// ---------------------------------------------------------------------------
__global__ __launch_bounds__(256) void gemm_wmma_generic_kernel(
    const float* __restrict__ Ag, const float* __restrict__ Wg,
    const float* __restrict__ bias, float* __restrict__ Cg,
    int M, int Nn, int K, int lda, int act)
{
  __shared__ __align__(16) _Float16 sA[64*32];
  __shared__ __align__(16) _Float16 sB[64*32];
  const int tid  = threadIdx.x;
  const int wave = tid >> 5;
  const int lane = tid & 31;
  const int bm = blockIdx.x * 64;
  const int bn = blockIdx.y * 64;
  const int wm = (wave & 3) * 16;
  const int wn = (wave >> 2) * 32;
  v8f acc0 = {0.f,0.f,0.f,0.f,0.f,0.f,0.f,0.f};
  v8f acc1 = acc0;

  for (int k0 = 0; k0 < K; k0 += 32) {
    for (int i = tid; i < 64*32; i += 256) {
      int r = i >> 5, c = i & 31;
      int gr = bm + r, gc = k0 + c;
      float v = 0.f;
      if (gr < M && gc < K) v = Ag[(size_t)gr*lda + gc];
      sA[i] = (_Float16)v;
    }
    for (int i = tid; i < 64*32; i += 256) {
      int r = i >> 5, c = i & 31;
      int gn = bn + r, gc = k0 + c;
      float v = 0.f;
      if (gn < Nn && gc < K) v = Wg[(size_t)gn*K + gc];
      sB[i] = (_Float16)v;
    }
    __syncthreads();
    v16h af, bf0, bf1;
    const int half8 = (lane >> 4) << 3;
    const int l16   = lane & 15;
    #pragma unroll
    for (int j = 0; j < 16; ++j) {
      int kk = half8 + (j & 7) + ((j >> 3) << 4);
      af[j]  = sA[(wm      + l16)*32 + kk];
      bf0[j] = sB[(wn      + l16)*32 + kk];
      bf1[j] = sB[(wn + 16 + l16)*32 + kk];
    }
    acc0 = __builtin_amdgcn_wmma_f32_16x16x32_f16(false, af, false, bf0, (short)0, acc0, false, false);
    acc1 = __builtin_amdgcn_wmma_f32_16x16x32_f16(false, af, false, bf1, (short)0, acc1, false, false);
    __syncthreads();
  }
  const int rb   = bm + wm + ((lane >> 4) << 3);
  const int col0 = bn + wn + (lane & 15);
  const int col1 = col0 + 16;
  float bv0 = 0.f, bv1 = 0.f;
  if (bias) { if (col0 < Nn) bv0 = bias[col0]; if (col1 < Nn) bv1 = bias[col1]; }
  #pragma unroll
  for (int r = 0; r < 8; ++r) {
    int row = rb + r;
    if (row < M) {
      if (col0 < Nn) Cg[(size_t)row*Nn + col0] = activate(acc0[r] + bv0, act);
      if (col1 < Nn) Cg[(size_t)row*Nn + col1] = activate(acc1[r] + bv1, act);
    }
  }
}

// ---------------------------------------------------------------------------
// Farthest point sampling: one block per batch, dists + argmax in LDS.
// ---------------------------------------------------------------------------
__global__ __launch_bounds__(256) void fps_kernel(const float* __restrict__ pts,
                                                  float* __restrict__ center)
{
  const int b = blockIdx.x, tid = threadIdx.x;
  __shared__ float dist[Np];
  __shared__ float rv[256];
  __shared__ int   ri[256];
  __shared__ int   last;
  for (int i = tid; i < Np; i += 256) dist[i] = 1e10f;
  if (tid == 0) {
    last = 0;
    center[((size_t)b*Gg)*3 + 0] = pts[(size_t)b*Np*3 + 0];
    center[((size_t)b*Gg)*3 + 1] = pts[(size_t)b*Np*3 + 1];
    center[((size_t)b*Gg)*3 + 2] = pts[(size_t)b*Np*3 + 2];
  }
  __syncthreads();
  for (int g = 1; g < Gg; ++g) {
    const int lp = last;
    const float lx = pts[((size_t)b*Np + lp)*3 + 0];
    const float ly = pts[((size_t)b*Np + lp)*3 + 1];
    const float lz = pts[((size_t)b*Np + lp)*3 + 2];
    for (int i = tid; i < Np; i += 256) {
      float dx = pts[((size_t)b*Np + i)*3 + 0] - lx;
      float dy = pts[((size_t)b*Np + i)*3 + 1] - ly;
      float dz = pts[((size_t)b*Np + i)*3 + 2] - lz;
      dist[i] = fminf(dist[i], dx*dx + dy*dy + dz*dz);
    }
    __syncthreads();
    float best = -1.f; int bi = 0;
    for (int i = tid; i < Np; i += 256)
      if (dist[i] > best) { best = dist[i]; bi = i; }
    rv[tid] = best; ri[tid] = bi;
    __syncthreads();
    for (int st = 128; st > 0; st >>= 1) {
      if (tid < st && rv[tid+st] > rv[tid]) { rv[tid] = rv[tid+st]; ri[tid] = ri[tid+st]; }
      __syncthreads();
    }
    if (tid == 0) {
      last = ri[0];
      center[((size_t)b*Gg + g)*3 + 0] = pts[((size_t)b*Np + ri[0])*3 + 0];
      center[((size_t)b*Gg + g)*3 + 1] = pts[((size_t)b*Np + ri[0])*3 + 1];
      center[((size_t)b*Gg + g)*3 + 2] = pts[((size_t)b*Np + ri[0])*3 + 2];
    }
    __syncthreads();
  }
}

// ---------------------------------------------------------------------------
// KNN (32 nearest) per (b,g): iterative argmin in LDS; neigh = pts - center.
// ---------------------------------------------------------------------------
__global__ __launch_bounds__(256) void knn_kernel(const float* __restrict__ pts,
                                                  const float* __restrict__ center,
                                                  float* __restrict__ neigh)
{
  const int bg = blockIdx.x, b = bg / Gg, tid = threadIdx.x;
  __shared__ float dist[Np];
  __shared__ float rv[256];
  __shared__ int   ri[256];
  const float cx = center[(size_t)bg*3 + 0];
  const float cy = center[(size_t)bg*3 + 1];
  const float cz = center[(size_t)bg*3 + 2];
  for (int i = tid; i < Np; i += 256) {
    float dx = pts[((size_t)b*Np + i)*3 + 0] - cx;
    float dy = pts[((size_t)b*Np + i)*3 + 1] - cy;
    float dz = pts[((size_t)b*Np + i)*3 + 2] - cz;
    dist[i] = dx*dx + dy*dy + dz*dz;
  }
  __syncthreads();
  for (int k = 0; k < GSs; ++k) {
    float best = 1e30f; int bi = 0;
    for (int i = tid; i < Np; i += 256)
      if (dist[i] < best) { best = dist[i]; bi = i; }
    rv[tid] = best; ri[tid] = bi;
    __syncthreads();
    for (int st = 128; st > 0; st >>= 1) {
      if (tid < st && rv[tid+st] < rv[tid]) { rv[tid] = rv[tid+st]; ri[tid] = ri[tid+st]; }
      __syncthreads();
    }
    if (tid == 0) {
      int idx = ri[0];
      neigh[((size_t)bg*GSs + k)*3 + 0] = pts[((size_t)b*Np + idx)*3 + 0] - cx;
      neigh[((size_t)bg*GSs + k)*3 + 1] = pts[((size_t)b*Np + idx)*3 + 1] - cy;
      neigh[((size_t)bg*GSs + k)*3 + 2] = pts[((size_t)b*Np + idx)*3 + 2] - cz;
      dist[idx] = 1e30f;
    }
    __syncthreads();
  }
}

// ---- per-column mean/var over M rows (point-encoder batch norm stats) ----
__global__ __launch_bounds__(256) void colstat_kernel(const float* __restrict__ x,
                                                      float* mu, float* var,
                                                      int Mrows, int C)
{
  const int c = blockIdx.x, tid = threadIdx.x;
  __shared__ float s1[256], s2[256];
  float a = 0.f, q = 0.f;
  for (int r = tid; r < Mrows; r += 256) {
    float v = x[(size_t)r*C + c]; a += v; q += v*v;
  }
  s1[tid] = a; s2[tid] = q;
  __syncthreads();
  for (int st = 128; st > 0; st >>= 1) {
    if (tid < st) { s1[tid] += s1[tid+st]; s2[tid] += s2[tid+st]; }
    __syncthreads();
  }
  if (tid == 0) {
    float m = s1[0] / (float)Mrows;
    mu[c] = m; var[c] = s2[0] / (float)Mrows - m*m;
  }
}

__global__ void bnrelu_kernel(float* x, const float* mu, const float* var,
                              const float* g, const float* be, size_t n, int C)
{
  size_t t = (size_t)blockIdx.x * blockDim.x + threadIdx.x;
  if (t >= n) return;
  int c = (int)(t % C);
  float v = (x[t] - mu[c]) * rsqrtf(var[c] + 1e-5f) * g[c] + be[c];
  x[t] = v > 0.f ? v : 0.f;
}

// ---- group max (256ch) + concat [max | f] -> 512ch ----
__global__ __launch_bounds__(256) void maxcat_kernel(const float* __restrict__ f2,
                                                     float* __restrict__ fcat)
{
  const int bg = blockIdx.x, c = threadIdx.x;
  float m = -1e30f;
  for (int k = 0; k < GSs; ++k) m = fmaxf(m, f2[((size_t)bg*GSs + k)*256 + c]);
  for (int k = 0; k < GSs; ++k) {
    size_t r = (size_t)bg*GSs + k;
    fcat[r*512 + c]       = m;
    fcat[r*512 + 256 + c] = f2[r*256 + c];
  }
}

__global__ __launch_bounds__(384) void maxpool384_kernel(const float* __restrict__ f,
                                                         float* __restrict__ tok)
{
  const int bg = blockIdx.x, c = threadIdx.x;
  float m = -1e30f;
  for (int k = 0; k < GSs; ++k) m = fmaxf(m, f[((size_t)bg*GSs + k)*DM + c]);
  tok[(size_t)bg*DM + c] = m;
}

// ---- stable ascending argsort of 128 values via rank counting ----
__global__ __launch_bounds__(128) void argsort_kernel(const float* __restrict__ noise,
                                                      int* mask_idx, int* vis_idx)
{
  const int b = blockIdx.x, tid = threadIdx.x;
  __shared__ float v[128];
  __shared__ int perm[128];
  v[tid] = noise[b*128 + tid];
  __syncthreads();
  float mv = v[tid]; int r = 0;
  for (int j = 0; j < 128; ++j) {
    float o = v[j];
    r += (o < mv) || (o == mv && j < tid);
  }
  perm[r] = tid;
  __syncthreads();
  if (tid < NMASK) mask_idx[b*NMASK + tid] = perm[tid];
  else             vis_idx[b*NVIS + (tid - NMASK)] = perm[tid];
}

// ---- elementwise helpers ----
__global__ void zero_kernel(float* p, size_t n){
  size_t t = (size_t)blockIdx.x*blockDim.x + threadIdx.x; if (t < n) p[t] = 0.f;
}
__global__ void add2_kernel(const float* a, const float* b, float* o, size_t n){
  size_t t = (size_t)blockIdx.x*blockDim.x + threadIdx.x; if (t < n) o[t] = a[t] + b[t];
}
__global__ void addinp_kernel(float* dst, const float* src, size_t n){
  size_t t = (size_t)blockIdx.x*blockDim.x + threadIdx.x; if (t < n) dst[t] += src[t];
}

__global__ void gather_rows_kernel(float* dst, const float* src, const int* idx,
                                   int Bn, int nIdx, int C, int srcRows)
{
  int t = blockIdx.x*blockDim.x + threadIdx.x;
  int tot = Bn*nIdx*C;
  if (t >= tot) return;
  int c = t % C, i = (t / C) % nIdx, b = t / (C*nIdx);
  dst[t] = src[((size_t)b*srcRows + idx[b*nIdx + i])*C + c];
}
__global__ void copy_rows_kernel(float* dst, const float* src, int Bn, int nRows,
                                 int C, int dstTot, int rowOff)
{
  int t = blockIdx.x*blockDim.x + threadIdx.x;
  int tot = Bn*nRows*C;
  if (t >= tot) return;
  int c = t % C, i = (t / C) % nRows, b = t / (C*nRows);
  dst[((size_t)b*dstTot + rowOff + i)*C + c] = src[t];
}
__global__ void extract_rows_kernel(float* dst, const float* src, int Bn, int nRows,
                                    int C, int srcTot, int rowOff)
{
  int t = blockIdx.x*blockDim.x + threadIdx.x;
  int tot = Bn*nRows*C;
  if (t >= tot) return;
  int c = t % C, i = (t / C) % nRows, b = t / (C*nRows);
  dst[t] = src[((size_t)b*srcTot + rowOff + i)*C + c];
}
__global__ void bcast_rows_kernel(float* dst, const float* tokv, int Bn, int nRows,
                                  int C, int dstTot, int rowOff)
{
  int t = blockIdx.x*blockDim.x + threadIdx.x;
  int tot = Bn*nRows*C;
  if (t >= tot) return;
  int c = t % C, i = (t / C) % nRows, b = t / (C*nRows);
  dst[((size_t)b*dstTot + rowOff + i)*C + c] = tokv[c];
}

// ---- layernorm over last dim 384; 128 threads/row; safe in-place ----
__global__ __launch_bounds__(128) void ln_kernel(const float* __restrict__ x,
                                                 float* __restrict__ y,
                                                 const float* __restrict__ w,
                                                 const float* __restrict__ bb,
                                                 int rows)
{
  const int r = blockIdx.x;
  if (r >= rows) return;
  const int tid = threadIdx.x;
  __shared__ float s[128];
  float v0 = x[(size_t)r*DM + tid];
  float v1 = x[(size_t)r*DM + tid + 128];
  float v2 = x[(size_t)r*DM + tid + 256];
  s[tid] = v0 + v1 + v2;
  __syncthreads();
  for (int st = 64; st > 0; st >>= 1) { if (tid < st) s[tid] += s[tid+st]; __syncthreads(); }
  float mean = s[0] / (float)DM;
  __syncthreads();
  float d0 = v0 - mean, d1 = v1 - mean, d2 = v2 - mean;
  s[tid] = d0*d0 + d1*d1 + d2*d2;
  __syncthreads();
  for (int st = 64; st > 0; st >>= 1) { if (tid < st) s[tid] += s[tid+st]; __syncthreads(); }
  float rstd = rsqrtf(s[0] / (float)DM + 1e-5f);
  y[(size_t)r*DM + tid]       = d0*rstd*w[tid]       + bb[tid];
  y[(size_t)r*DM + tid + 128] = d1*rstd*w[tid + 128] + bb[tid + 128];
  y[(size_t)r*DM + tid + 256] = d2*rstd*w[tid + 256] + bb[tid + 256];
}

// ---- causal depthwise conv1d (k=4, pad 3 left) + SiLU.  xi lives in xz[:, :768]
__global__ void conv_silu_kernel(const float* __restrict__ xz,
                                 const float* __restrict__ cw,
                                 const float* __restrict__ cb,
                                 float* __restrict__ xc, int Bn, int L)
{
  int t = blockIdx.x*blockDim.x + threadIdx.x;
  if (t >= Bn*DI) return;
  const int b = t / DI, d = t % DI;
  const float w0 = cw[d*4], w1 = cw[d*4+1], w2 = cw[d*4+2], w3 = cw[d*4+3];
  const float bv = cb[d];
  for (int l = 0; l < L; ++l) {
    size_t base = (size_t)(b*L + l)*(2*DI) + d;
    float acc = bv + w3*xz[base];
    if (l >= 1) acc += w2*xz[base - (size_t)(2*DI)];
    if (l >= 2) acc += w1*xz[base - (size_t)(4*DI)];
    if (l >= 3) acc += w0*xz[base - (size_t)(6*DI)];
    xc[(size_t)(b*L + l)*DI + d] = acc * sigf(acc);
  }
}

// ---- selective scan: thread = (b, d_inner), 16 states in registers; fused
//      D-skip and SiLU(z) gating (z lives in xz[:, 768:]).
__global__ void scan_kernel(const float* __restrict__ dt,
                            const float* __restrict__ xc,
                            const float* __restrict__ dbc,
                            const float* __restrict__ xz,
                            const float* __restrict__ A_log,
                            const float* __restrict__ Dp,
                            float* __restrict__ yg, int Bn, int L)
{
  int t = blockIdx.x*blockDim.x + threadIdx.x;
  if (t >= Bn*DI) return;
  const int b = t / DI, d = t % DI;
  float a[DSt], h[DSt];
  #pragma unroll
  for (int s = 0; s < DSt; ++s) { a[s] = -expf(A_log[d*DSt + s]); h[s] = 0.f; }
  const float dk = Dp[d];
  for (int l = 0; l < L; ++l) {
    size_t row = (size_t)(b*L + l);
    float dtv = dt[row*DI + d];
    float xv  = xc[row*DI + d];
    const float* br = dbc + row*56 + DTR;
    const float* cr = br + DSt;
    float y = 0.f;
    #pragma unroll
    for (int s = 0; s < DSt; ++s) {
      h[s] = h[s]*expf(dtv*a[s]) + dtv*xv*br[s];
      y += h[s]*cr[s];
    }
    y += dk*xv;
    float zv = xz[row*(2*DI) + DI + d];
    yg[row*DI + d] = y * (zv * sigf(zv));
  }
}

// ---- chamfer over one (32 x 32) pair per block ----
__global__ __launch_bounds__(32) void chamfer_kernel(const float* __restrict__ x,
                                                     const float* __restrict__ y,
                                                     float* __restrict__ accum)
{
  const int p = blockIdx.x, t = threadIdx.x;
  __shared__ float dsh[GSs*GSs];
  __shared__ float rmin[GSs], cmin[GSs];
  const float* xp = x + (size_t)p*GSs*3;
  const float* yp = y + (size_t)p*GSs*3;
  float ax = xp[t*3], ay = xp[t*3+1], az = xp[t*3+2];
  for (int j = 0; j < GSs; ++j) {
    float dx = ax - yp[j*3], dy = ay - yp[j*3+1], dz = az - yp[j*3+2];
    dsh[t*GSs + j] = dx*dx + dy*dy + dz*dz;
  }
  __syncthreads();
  float rm = 1e30f, cm = 1e30f;
  for (int j = 0; j < GSs; ++j) {
    rm = fminf(rm, dsh[t*GSs + j]);
    cm = fminf(cm, dsh[j*GSs + t]);
  }
  rmin[t] = rm; cmin[t] = cm;
  __syncthreads();
  if (t == 0) {
    float s0 = 0.f, s1 = 0.f;
    for (int j = 0; j < GSs; ++j) { s0 += rmin[j]; s1 += cmin[j]; }
    atomicAdd(&accum[0], s0);
    atomicAdd(&accum[1], s1);
  }
}

__global__ void final_kernel(const float* accum, float* out){
  out[0] = (accum[0] + accum[1]) / (float)(Bb*NMASK*GSs);
}

// ===========================================================================
// Host side
// ===========================================================================
static void launch_gemm(const float* A, const float* W, const float* bias, float* C,
                        int M, int Nn, int K, int lda, int act, hipStream_t s)
{
  if ((K & 31) == 0) {
    dim3 grid(CEILDIV(M, 64), CEILDIV(Nn, 128));
    gemm_wmma_aligned_kernel<<<grid, 256, 0, s>>>(A, W, bias, C, M, Nn, K, lda, act);
  } else {
    dim3 grid(CEILDIV(M, 64), CEILDIV(Nn, 64));
    gemm_wmma_generic_kernel<<<grid, 256, 0, s>>>(A, W, bias, C, M, Nn, K, lda, act);
  }
}

static void run_posmlp(const float* c, const float* w1, const float* b1,
                       const float* w2, const float* b2, float* hidden, float* out,
                       int M, hipStream_t s)
{
  launch_gemm(c, w1, b1, hidden, M, 128, 3, 3, /*gelu*/2, s);
  launch_gemm(hidden, w2, b2, out, M, DM, 128, 128, 0, s);
}

static void run_mixer(const float* tok, const float* pos, int Bn, int L, int depth,
                      const float* norm_w, const float* norm_b, const float* in_w,
                      const float* conv_w, const float* conv_b, const float* xp_w,
                      const float* dt_w, const float* dt_b, const float* A_log,
                      const float* Dp, const float* out_w,
                      const float* nf_w, const float* nf_b,
                      float* h, float* res, float* hn, float* xz, float* xc,
                      float* dbc, float* dtb, float* yg, float* out, hipStream_t s)
{
  const int M = Bn * L;
  const size_t n = (size_t)M * DM;
  add2_kernel<<<CEILDIV(n,(size_t)256),256,0,s>>>(tok, pos, h, n);
  zero_kernel<<<CEILDIV(n,(size_t)256),256,0,s>>>(res, n);
  for (int l = 0; l < depth; ++l) {
    addinp_kernel<<<CEILDIV(n,(size_t)256),256,0,s>>>(res, h, n);        // res += h
    ln_kernel<<<M,128,0,s>>>(res, hn, norm_w + (size_t)l*DM, norm_b + (size_t)l*DM, M);
    launch_gemm(hn, in_w + (size_t)l*2*DI*DM, nullptr, xz, M, 2*DI, DM, DM, 0, s);
    conv_silu_kernel<<<CEILDIV(Bn*DI,256),256,0,s>>>(xz, conv_w + (size_t)l*DI*4,
                                                     conv_b + (size_t)l*DI, xc, Bn, L);
    launch_gemm(xc, xp_w + (size_t)l*56*DI, nullptr, dbc, M, 56, DI, DI, 0, s);
    launch_gemm(dbc, dt_w + (size_t)l*DI*DTR, dt_b + (size_t)l*DI, dtb,
                M, DI, DTR, 56, /*softplus*/3, s);
    scan_kernel<<<CEILDIV(Bn*DI,256),256,0,s>>>(dtb, xc, dbc, xz,
                                                A_log + (size_t)l*DI*DSt,
                                                Dp + (size_t)l*DI, yg, Bn, L);
    launch_gemm(yg, out_w + (size_t)l*DM*DI, nullptr, h, M, DM, DI, DI, 0, s);
  }
  addinp_kernel<<<CEILDIV(n,(size_t)256),256,0,s>>>(res, h, n);          // res = h + res
  ln_kernel<<<M,128,0,s>>>(res, out, nf_w, nf_b, M);
}

extern "C" void kernel_launch(void* const* d_in, const int* in_sizes, int n_in,
                              void* d_out, int out_size, void* d_ws, size_t ws_size,
                              hipStream_t stream)
{
  (void)in_sizes; (void)n_in; (void)out_size; (void)ws_size;
  const float* pts     = (const float*)d_in[0];
  const float* noise   = (const float*)d_in[1];
  const float* pe_w1   = (const float*)d_in[2];
  const float* pe_b1   = (const float*)d_in[3];
  const float* pe_g1   = (const float*)d_in[4];
  const float* pe_be1  = (const float*)d_in[5];
  const float* pe_w2   = (const float*)d_in[6];
  const float* pe_b2   = (const float*)d_in[7];
  const float* pe_w3   = (const float*)d_in[8];
  const float* pe_b3   = (const float*)d_in[9];
  const float* pe_g2   = (const float*)d_in[10];
  const float* pe_be2  = (const float*)d_in[11];
  const float* pe_w4   = (const float*)d_in[12];
  const float* pe_b4   = (const float*)d_in[13];
  const float* ep_w1   = (const float*)d_in[14];
  const float* ep_b1   = (const float*)d_in[15];
  const float* ep_w2   = (const float*)d_in[16];
  const float* ep_b2   = (const float*)d_in[17];
  const float* dp_w1   = (const float*)d_in[18];
  const float* dp_b1   = (const float*)d_in[19];
  const float* dp_w2   = (const float*)d_in[20];
  const float* dp_b2   = (const float*)d_in[21];
  const float* mask_tok= (const float*)d_in[22];
  const float* inc_w   = (const float*)d_in[23];
  const float* inc_b   = (const float*)d_in[24];
  const float* e_norm_w= (const float*)d_in[25];
  const float* e_norm_b= (const float*)d_in[26];
  const float* e_in_w  = (const float*)d_in[27];
  const float* e_conv_w= (const float*)d_in[28];
  const float* e_conv_b= (const float*)d_in[29];
  const float* e_xp_w  = (const float*)d_in[30];
  const float* e_dt_w  = (const float*)d_in[31];
  const float* e_dt_b  = (const float*)d_in[32];
  const float* e_A_log = (const float*)d_in[33];
  const float* e_D     = (const float*)d_in[34];
  const float* e_out_w = (const float*)d_in[35];
  const float* d_norm_w= (const float*)d_in[36];
  const float* d_norm_b= (const float*)d_in[37];
  const float* d_in_w  = (const float*)d_in[38];
  const float* d_conv_w= (const float*)d_in[39];
  const float* d_conv_b= (const float*)d_in[40];
  const float* d_xp_w  = (const float*)d_in[41];
  const float* d_dt_w  = (const float*)d_in[42];
  const float* d_dt_b  = (const float*)d_in[43];
  const float* d_A_log = (const float*)d_in[44];
  const float* d_Dp    = (const float*)d_in[45];
  const float* d_out_w = (const float*)d_in[46];
  const float* e_nf_w  = (const float*)d_in[47];
  const float* d_nf_w  = (const float*)d_in[48];
  const float* dec_norm_w = (const float*)d_in[49];
  const float* e_nf_b  = (const float*)d_in[50];
  const float* d_nf_b  = (const float*)d_in[51];
  const float* dec_norm_b = (const float*)d_in[52];
  float* outp = (float*)d_out;

  // ------------- workspace carve-out -------------
  float* wsf = (float*)d_ws;
  size_t off = 0;
  auto A = [&](size_t nelem)->float* { float* p = wsf + off; off += nelem; return p; };
  float* center  = A((size_t)Bb*Gg*3);
  float* neigh   = A((size_t)Bb*Gg*GSs*3);
  float* R1      = A((size_t)65536*512);       // ping
  float* R2      = A((size_t)65536*512);       // pong
  float* mu      = A(512);
  float* var     = A(512);
  float* tokens  = A((size_t)Bb*Gg*DM);
  int*   mask_idx= (int*)A(Bb*NMASK);
  int*   vis_idx = (int*)A(Bb*NVIS);
  float* tok_vis = A((size_t)Bb*NVIS*DM);
  float* c_vis   = A((size_t)Bb*NVIS*3);
  float* c_mask  = A((size_t)Bb*NMASK*3);
  float* c_full  = A((size_t)Bb*Gg*3);
  float* posh    = A((size_t)Bb*Gg*128);
  float* pos_vis = A((size_t)Bb*NVIS*DM);
  float* pos_full= A((size_t)Bb*Gg*DM);
  float* hbuf    = A((size_t)Bb*Gg*DM);
  float* resbuf  = A((size_t)Bb*Gg*DM);
  float* hnbuf   = A((size_t)Bb*Gg*DM);
  float* xzbuf   = A((size_t)Bb*Gg*2*DI);
  float* xcbuf   = A((size_t)Bb*Gg*DI);
  float* dbcbuf  = A((size_t)Bb*Gg*56);
  float* dtbuf   = A((size_t)Bb*Gg*DI);
  float* ygbuf   = A((size_t)Bb*Gg*DI);
  float* xvis    = A((size_t)Bb*NVIS*DM);
  float* xfull   = A((size_t)Bb*Gg*DM);
  float* xrec    = A((size_t)Bb*Gg*DM);
  float* xm      = A((size_t)Bb*NMASK*DM);
  float* rebuild = A((size_t)Bb*NMASK*GSs*3);
  float* gt      = A((size_t)Bb*NMASK*GSs*3);
  float* accum   = A(2);

  // ------------- grouping -------------
  fps_kernel<<<Bb, 256, 0, stream>>>(pts, center);
  knn_kernel<<<Bb*Gg, 256, 0, stream>>>(pts, center, neigh);

  // ------------- point encoder -------------
  const int Mpe = Bb*Gg*GSs;  // 65536
  launch_gemm(neigh, pe_w1, pe_b1, R1, Mpe, 128, 3, 3, 0, stream);
  colstat_kernel<<<128, 256, 0, stream>>>(R1, mu, var, Mpe, 128);
  bnrelu_kernel<<<CEILDIV((size_t)Mpe*128,(size_t)256),256,0,stream>>>(R1, mu, var, pe_g1, pe_be1, (size_t)Mpe*128, 128);
  launch_gemm(R1, pe_w2, pe_b2, R2, Mpe, 256, 128, 128, 0, stream);
  maxcat_kernel<<<Bb*Gg, 256, 0, stream>>>(R2, R1);                       // R1 = [max|f] 512
  launch_gemm(R1, pe_w3, pe_b3, R2, Mpe, 512, 512, 512, 0, stream);
  colstat_kernel<<<512, 256, 0, stream>>>(R2, mu, var, Mpe, 512);
  bnrelu_kernel<<<CEILDIV((size_t)Mpe*512,(size_t)256),256,0,stream>>>(R2, mu, var, pe_g2, pe_be2, (size_t)Mpe*512, 512);
  launch_gemm(R2, pe_w4, pe_b4, R1, Mpe, DM, 512, 512, 0, stream);
  maxpool384_kernel<<<Bb*Gg, 384, 0, stream>>>(R1, tokens);

  // ------------- masking -------------
  argsort_kernel<<<Bb, 128, 0, stream>>>(noise, mask_idx, vis_idx);
  {
    int tot = Bb*NVIS*DM;
    gather_rows_kernel<<<CEILDIV(tot,256),256,0,stream>>>(tok_vis, tokens, vis_idx, Bb, NVIS, DM, Gg);
    tot = Bb*NVIS*3;
    gather_rows_kernel<<<CEILDIV(tot,256),256,0,stream>>>(c_vis, center, vis_idx, Bb, NVIS, 3, Gg);
    tot = Bb*NMASK*3;
    gather_rows_kernel<<<CEILDIV(tot,256),256,0,stream>>>(c_mask, center, mask_idx, Bb, NMASK, 3, Gg);
  }

  // ------------- encoder -------------
  run_posmlp(c_vis, ep_w1, ep_b1, ep_w2, ep_b2, posh, pos_vis, Bb*NVIS, stream);
  run_mixer(tok_vis, pos_vis, Bb, NVIS, ENCD,
            e_norm_w, e_norm_b, e_in_w, e_conv_w, e_conv_b, e_xp_w,
            e_dt_w, e_dt_b, e_A_log, e_D, e_out_w, e_nf_w, e_nf_b,
            hbuf, resbuf, hnbuf, xzbuf, xcbuf, dbcbuf, dtbuf, ygbuf, xvis, stream);

  // ------------- decoder inputs -------------
  copy_rows_kernel<<<CEILDIV(Bb*NVIS*3,256),256,0,stream>>>(c_full, c_vis, Bb, NVIS, 3, Gg, 0);
  copy_rows_kernel<<<CEILDIV(Bb*NMASK*3,256),256,0,stream>>>(c_full, c_mask, Bb, NMASK, 3, Gg, NVIS);
  run_posmlp(c_full, dp_w1, dp_b1, dp_w2, dp_b2, posh, pos_full, Bb*Gg, stream);
  copy_rows_kernel<<<CEILDIV(Bb*NVIS*DM,256),256,0,stream>>>(xfull, xvis, Bb, NVIS, DM, Gg, 0);
  bcast_rows_kernel<<<CEILDIV(Bb*NMASK*DM,256),256,0,stream>>>(xfull, mask_tok, Bb, NMASK, DM, Gg, NVIS);

  // ------------- decoder -------------
  run_mixer(xfull, pos_full, Bb, Gg, DECD,
            d_norm_w, d_norm_b, d_in_w, d_conv_w, d_conv_b, d_xp_w,
            d_dt_w, d_dt_b, d_A_log, d_Dp, d_out_w, d_nf_w, d_nf_b,
            hbuf, resbuf, hnbuf, xzbuf, xcbuf, dbcbuf, dtbuf, ygbuf, xrec, stream);

  // ------------- reconstruction head -------------
  extract_rows_kernel<<<CEILDIV(Bb*NMASK*DM,256),256,0,stream>>>(xm, xrec, Bb, NMASK, DM, Gg, NVIS);
  ln_kernel<<<Bb*NMASK, 128, 0, stream>>>(xm, xm, dec_norm_w, dec_norm_b, Bb*NMASK);
  launch_gemm(xm, inc_w, inc_b, rebuild, Bb*NMASK, 3*GSs, DM, DM, 0, stream);
  gather_rows_kernel<<<CEILDIV(Bb*NMASK*GSs*3,256),256,0,stream>>>(gt, neigh, mask_idx, Bb, NMASK, GSs*3, Gg);

  // ------------- chamfer -------------
  zero_kernel<<<1, 2, 0, stream>>>(accum, 2);
  chamfer_kernel<<<Bb*NMASK, 32, 0, stream>>>(rebuild, gt, accum);
  final_kernel<<<1, 1, 0, stream>>>(accum, outp);
}
